// NeighborAttention_54692113547876
// MI455X (gfx1250) — matrix operand order repaired and checked by reading, not compiled, not run on hardware
//
#include <hip/hip_runtime.h>

// ---- CDNA5 (gfx1250) neighbor attention -------------------------------------
// Memory-bound on h_E (768 MB @ 23.3 TB/s ~= 33us). KV projection runs on
// v_wmma_f32_16x16x32_f16 (f16 in, f32 accumulate). Weights pre-swizzled to
// B-fragment layout in d_ws by a tiny pre-kernel. Epilogue (softmax over 32
// neighbors, ctx, W_O) is scalar VALU with wave32 shuffle reductions.

typedef __attribute__((ext_vector_type(16))) _Float16 v16h;
typedef __attribute__((ext_vector_type(8)))  _Float16 v8h;
typedef __attribute__((ext_vector_type(8)))  float    v8f;

#define NH     128            // NUM_HIDDEN
#define NIN    384            // NUM_IN
#define KNB    32             // neighbors per site
#define DH     32             // head dim
#define SITES  2              // sites per workgroup
#define AROWS  (SITES * KNB)  // 64 A-matrix rows
#define KSTEPS (NIN / 32)     // 12 WMMA K-steps
#define NTILES ((2 * NH) / 16)// 16 N-tiles (Kt || V concatenated: 256 cols)
#define SA_STR 392            // f16 LDS stride for A (384 + 8 pad: bank-safe)
#define SKV_STR 260           // f32 LDS stride for KV out (256 + 4 pad)

// LDS carve (bytes)
#define OFF_A    0
#define OFF_KV   (OFF_A  + AROWS * SA_STR * 2)      // 50176
#define OFF_Q    (OFF_KV + AROWS * SKV_STR * 4)     // +66560 = 116736
#define OFF_HV   (OFF_Q   + SITES * NH * 4)         // +1024
#define OFF_MASK (OFF_HV  + SITES * NH * 4)         // +1024
#define OFF_ATT  (OFF_MASK + SITES * KNB * 4)       // +256
#define OFF_CTX  (OFF_ATT + SITES * NH * 4)         // +1024
#define SMEM_SZ  (OFF_CTX + SITES * NH * 4)         // 121088 total

// ---- Pre-kernel: swizzle [W_K ; W_V] (256 x 384 f32) into f16 B-fragments ---
// B-matrix 32x16 f16, contiguous-K order used here (same as A): lane n (0-15) holds K=0..15 in V0..7
// (2 f16/dword), lane n+16 holds K=16..31.  Flat dword index:
//   ((nt*KSTEPS + ks)*32 + lane)*8 + j   ->  one contiguous 32B load per lane.
__global__ void wkv_swizzle_kernel(const float* __restrict__ W_K,
                                   const float* __restrict__ W_V,
                                   unsigned* __restrict__ wf) {
    int flat = blockIdx.x * 256 + threadIdx.x;  // dword index, 49152 total
    int j    = flat & 7;
    int t2   = flat >> 3;
    int lane = t2 & 31;
    int t3   = t2 >> 5;
    int ks   = t3 % KSTEPS;
    int nt   = t3 / KSTEPS;
    int n    = nt * 16 + (lane & 15);
    int kk   = ks * 32 + ((lane < 16) ? 0 : 16) + 2 * j;
    const float* src = (n < NH) ? (W_K + n * NIN + kk)
                                : (W_V + (n - NH) * NIN + kk);
    union { _Float16 h[2]; unsigned u; } p;
    p.h[0] = (_Float16)src[0];
    p.h[1] = (_Float16)src[1];
    wf[flat] = p.u;
}

// ---- Main kernel: one workgroup = 2 sites -----------------------------------
__launch_bounds__(256)
__global__ void neigh_attn_kernel(const float* __restrict__ h_V,
                                  const float* __restrict__ h_E,
                                  const float* __restrict__ mask,
                                  const float* __restrict__ W_Q,
                                  const float* __restrict__ W_O,
                                  const unsigned* __restrict__ wf,
                                  float* __restrict__ out) {
    extern __shared__ char smem[];
    _Float16* sA   = (_Float16*)(smem + OFF_A);
    float*    sKV  = (float*)(smem + OFF_KV);
    float*    sQ   = (float*)(smem + OFF_Q);
    float*    sHV  = (float*)(smem + OFF_HV);
    float*    sMsk = (float*)(smem + OFF_MASK);
    float*    sAtt = (float*)(smem + OFF_ATT);
    float*    sCtx = (float*)(smem + OFF_CTX);

    const int tid   = threadIdx.x;
    const int lane  = tid & 31;
    const int wv    = tid >> 5;
    const long site0 = (long)blockIdx.x * SITES;   // flat b*N + n

    // ---- stage h_V rows and mask ----
    sHV[tid] = h_V[site0 * NH + tid];
    if (tid < SITES * KNB) sMsk[tid] = mask[site0 * KNB + tid];

    // ---- stage h_E tile (64 x 384) f32 -> f16 LDS, float4 coalesced ----
    {
        const float4* src4 = (const float4*)(h_E + site0 * (KNB * NIN));
        #pragma unroll
        for (int f = tid; f < (AROWS * NIN) / 4; f += 256) {
            int i = f * 4, row = i / NIN, col = i - row * NIN;
            float4 v = src4[f];
            _Float16* d = sA + row * SA_STR + col;
            d[0] = (_Float16)v.x; d[1] = (_Float16)v.y;
            d[2] = (_Float16)v.z; d[3] = (_Float16)v.w;
        }
    }
    __syncthreads();

    // ---- Q = h_V @ W_Q^T (one output per thread, f32) ----
    {
        int s = tid >> 7, i = tid & 127;
        const float4* hv4 = (const float4*)(sHV + s * NH);
        const float4* wq4 = (const float4*)(W_Q + i * NH);
        float acc = 0.f;
        #pragma unroll 8
        for (int j = 0; j < NH / 4; ++j) {
            float4 a = hv4[j], b = wq4[j];
            acc += a.x * b.x + a.y * b.y + a.z * b.z + a.w * b.w;
        }
        sQ[tid] = acc;
    }

    // ---- WMMA: (64 x 384) f16 @ (384 x 256) f16 -> (64 x 256) f32 ----
    // wave w owns nt = {2w, 2w+1}, all 4 M-tiles.
    v8f acc[8] = {};
    const int nt0 = wv * 2, nt1 = wv * 2 + 1;
    const int kbaseA = (lane < 16) ? 0 : 16;    // A-frag K-base, contiguous K (matches wf B-frags)
    const int arow = lane & 15;
    #pragma unroll 4
    for (int ks = 0; ks < KSTEPS; ++ks) {
        v16h Afr[4];
        #pragma unroll
        for (int mt = 0; mt < 4; ++mt) {
            const _Float16* p = sA + (mt * 16 + arow) * SA_STR + ks * 32 + kbaseA;
            v8h lo = *(const v8h*)p;            // K = kbase .. kbase+7
            v8h hi = *(const v8h*)(p + 8);      // K = kbase+8 .. kbase+15
            Afr[mt] = __builtin_shufflevector(lo, hi,
                0,1,2,3,4,5,6,7,8,9,10,11,12,13,14,15);
        }
        v16h B0 = *(const v16h*)(wf + (((long)nt0 * KSTEPS + ks) * 32 + lane) * 8);
        v16h B1 = *(const v16h*)(wf + (((long)nt1 * KSTEPS + ks) * 32 + lane) * 8);
        #pragma unroll
        for (int mt = 0; mt < 4; ++mt) {
            acc[mt]     = __builtin_amdgcn_wmma_f32_16x16x32_f16(
                false, Afr[mt], false, B0, (short)0, acc[mt],     false, false);
            acc[4 + mt] = __builtin_amdgcn_wmma_f32_16x16x32_f16(
                false, Afr[mt], false, B1, (short)0, acc[4 + mt], false, false);
        }
    }
    // D layout: VGPR j -> M = j (lanes 0-15) / j+8 (lanes 16-31), N = lane&15
    {
        int hi8 = (lane >= 16) ? 8 : 0;
        int col0 = nt0 * 16 + arow, col1 = nt1 * 16 + arow;
        #pragma unroll
        for (int mt = 0; mt < 4; ++mt)
            #pragma unroll
            for (int j = 0; j < 8; ++j) {
                sKV[(mt * 16 + hi8 + j) * SKV_STR + col0] = acc[mt][j];
                sKV[(mt * 16 + hi8 + j) * SKV_STR + col1] = acc[4 + mt][j];
            }
    }
    __syncthreads();

    // ---- logits + masked softmax over 32 neighbors (one wave per (site,head))
    {
        int s = tid >> 7, r = tid & 127, h = r >> 5, k = r & 31;
        const float* q  = sQ  + s * NH + h * DH;
        const float* kt = sKV + (s * KNB + k) * SKV_STR + h * DH;
        float lg = 0.f;
        #pragma unroll 8
        for (int d = 0; d < DH; ++d) lg += q[d] * kt[d];
        lg *= 0.17677669529663687f;                       // 1/sqrt(32)
        float mv = sMsk[s * KNB + k];
        lg = (mv > 0.f) ? lg : -3.4028234663852886e38f;   // finfo(f32).min
        float mx = lg;
        #pragma unroll
        for (int o = 16; o > 0; o >>= 1) mx = fmaxf(mx, __shfl_xor(mx, o, 32));
        float p = __expf(lg - mx);
        float sum = p;
        #pragma unroll
        for (int o = 16; o > 0; o >>= 1) sum += __shfl_xor(sum, o, 32);
        sAtt[tid] = (p / sum) * mv;
    }
    __syncthreads();

    // ---- ctx = attend @ V ----
    {
        int s = tid >> 7, r = tid & 127, h = r >> 5, d = r & 31;
        const float* at = sAtt + s * NH + h * DH;
        const float* vv = sKV + (s * KNB) * SKV_STR + NH + h * DH + d;
        float c = 0.f;
        #pragma unroll 8
        for (int k = 0; k < KNB; ++k) c += at[k] * vv[k * SKV_STR];
        sCtx[tid] = c;
    }
    __syncthreads();

    // ---- out = ctx @ W_O^T ----
    {
        int s = tid >> 7, i = tid & 127;
        const float4* c4 = (const float4*)(sCtx + s * NH);
        const float4* wo4 = (const float4*)(W_O + i * NH);
        float o = 0.f;
        #pragma unroll 8
        for (int j = 0; j < NH / 4; ++j) {
            float4 a = c4[j], b = wo4[j];
            o += a.x * b.x + a.y * b.y + a.z * b.z + a.w * b.w;
        }
        out[(site0 + s) * NH + i] = o;
    }
}

extern "C" void kernel_launch(void* const* d_in, const int* in_sizes, int n_in,
                              void* d_out, int out_size, void* d_ws, size_t ws_size,
                              hipStream_t stream) {
    const float* h_V  = (const float*)d_in[0];
    const float* h_E  = (const float*)d_in[1];
    const float* mask = (const float*)d_in[2];
    const float* W_Q  = (const float*)d_in[3];
    const float* W_K  = (const float*)d_in[4];
    const float* W_V  = (const float*)d_in[5];
    const float* W_O  = (const float*)d_in[6];
    float* out   = (float*)d_out;
    unsigned* wf = (unsigned*)d_ws;   // 192 KB of f16 B-fragments

    // 16*12*32*8 = 49152 dwords of swizzled weights
    wkv_swizzle_kernel<<<(NTILES * KSTEPS * 32 * 8) / 256, 256, 0, stream>>>(
        W_K, W_V, wf);

    int nsites = in_sizes[0] / NH;        // B*N = 16384
    int nblk = nsites / SITES;            // 8192 workgroups
    neigh_attn_kernel<<<nblk, 256, SMEM_SZ, stream>>>(
        h_V, h_E, mask, W_Q, W_O, wf, out);
}
